// Transformer_26448408608958
// MI455X (gfx1250) — matrix operand order repaired and checked by reading, not compiled, not run on hardware
//
#include <hip/hip_runtime.h>
#include <hip/hip_bf16.h>

typedef __attribute__((ext_vector_type(16))) __bf16 v16bf;
typedef __attribute__((ext_vector_type(8)))  __bf16 v8bf;
typedef __attribute__((ext_vector_type(8)))  float  v8f;
typedef __attribute__((ext_vector_type(4)))  float  v4f;

#define N_   32
#define NQ_  40
#define T_   4096
#define D_   256
#define H_   8
#define P_   4
#define DH_  32
#define K2D  512   // 2*D
#define BSTRIDE 40 // skewed LDS row stride in bf16 elems (80 B = 20 banks, conflict-free)

// ---------------------------------------------------------------------------
// Kernel 1: convert W_val (256x512) and W_out (256x256) fp32 -> bf16 (once).
// ---------------------------------------------------------------------------
__global__ __launch_bounds__(256) void cvt_weights(
    const float* __restrict__ wv, const float* __restrict__ wo,
    __bf16* __restrict__ wvb, __bf16* __restrict__ wob)
{
    int i = blockIdx.x * 256 + threadIdx.x;
    if (i < D_ * K2D) wvb[i] = (__bf16)wv[i];
    if (i < D_ * D_)  wob[i] = (__bf16)wo[i];
}

// ---------------------------------------------------------------------------
// Kernel 2: val = features @ W_val^T + b_val   (M=131072, K=512, N=256)
// One wave per 16-row M tile covering all 16 N tiles. bf16 WMMA, f32 accum.
// B slices double-buffered in LDS; the global->register phase of the fill is
// issued BEFORE the WMMA burst and the register->LDS commit AFTER it, so the
// burst contains only ds_load + wmma. A fp32 double-buffered in registers.
// ---------------------------------------------------------------------------
__device__ __forceinline__ void slice_gload(v8bf gv[4],
                                            const __bf16* __restrict__ wvb,
                                            int kk, int t)
{
    #pragma unroll
    for (int it = 0; it < 4; ++it) {
        const int c   = t + it * 256;      // 16-byte chunk id, 0..1023
        const int row = c >> 2;
        const int ko  = (c & 3) * 8;
        gv[it] = *(const v8bf*)(wvb + (size_t)row * K2D + kk * 32 + ko);
    }
}

__device__ __forceinline__ void slice_commit(__bf16* __restrict__ dst,
                                             const v8bf gv[4], int t)
{
    #pragma unroll
    for (int it = 0; it < 4; ++it) {
        const int c   = t + it * 256;
        const int row = c >> 2;
        const int ko  = (c & 3) * 8;
        *(v8bf*)(dst + row * BSTRIDE + ko) = gv[it];
    }
}

__global__ __launch_bounds__(256) void gemm_val(
    const float* __restrict__ feats, const __bf16* __restrict__ wvb,
    const float* __restrict__ bval, float* __restrict__ val)
{
    __shared__ __bf16 bsm[2][D_ * BSTRIDE];   // 2 x 20 KB

    const int t     = threadIdx.x;
    const int lane  = t & 31;
    const int lrow  = lane & 15;        // row within tile (A) / col (B,C)
    const int khalf = lane >> 4;        // K-half selector
    const int wid   = t >> 5;
    const int mtile = blockIdx.x * 8 + wid;      // 0..8191

    const float* arow = feats + ((size_t)mtile * 16 + lrow) * K2D;

    v8f acc[16] = {};
    v8bf gv[2][4];

    // prologue: slice 0 -> regs -> LDS buffer 0; load A(0)
    slice_gload(gv[0], wvb, 0, t);
    slice_commit(bsm[0], gv[0], t);
    v4f f0 = *(const v4f*)(arow + khalf * 8);
    v4f f1 = *(const v4f*)(arow + khalf * 8 + 4);
    v4f f2 = *(const v4f*)(arow + khalf * 8 + 16);
    v4f f3 = *(const v4f*)(arow + khalf * 8 + 20);

    for (int kk = 0; kk < 16; ++kk) {
        const int pn = (kk + 1) & 1;
        // issue next slice's global loads (in flight across barrier + burst)
        if (kk + 1 < 16) slice_gload(gv[pn], wvb, kk + 1, t);

        __syncthreads();                          // slice kk resident in LDS

        // issue next A loads early (dup of last slice when kk==15, harmless)
        const int kn = (kk + 1 < 16) ? kk + 1 : kk;
        const int kb = kn * 32 + khalf * 8;
        v4f g0 = *(const v4f*)(arow + kb);
        v4f g1 = *(const v4f*)(arow + kb + 4);
        v4f g2 = *(const v4f*)(arow + kb + 16);
        v4f g3 = *(const v4f*)(arow + kb + 20);

        v16bf a;
        #pragma unroll
        for (int i = 0; i < 4; ++i) {
            a[i]      = (__bf16)f0[i];
            a[4 + i]  = (__bf16)f1[i];
            a[8 + i]  = (__bf16)f2[i];
            a[12 + i] = (__bf16)f3[i];
        }

        const __bf16* bs = bsm[kk & 1] + lrow * BSTRIDE + khalf * 8;
        #pragma unroll
        for (int nt = 0; nt < 16; ++nt) {
            const __bf16* bp = bs + nt * 16 * BSTRIDE;
            v8bf b0 = *(const v8bf*)(bp);
            v8bf b1 = *(const v8bf*)(bp + 16);
            v16bf bfrag;
            #pragma unroll
            for (int i = 0; i < 8; ++i) { bfrag[i] = b0[i]; bfrag[8 + i] = b1[i]; }
            acc[nt] = __builtin_amdgcn_wmma_f32_16x16x32_bf16(
                false, a, false, bfrag, (short)0, acc[nt], false, false);
        }

        // commit next slice to the other LDS buffer (loadcnt now long satisfied)
        if (kk + 1 < 16) slice_commit(bsm[pn], gv[pn], t);

        f0 = g0; f1 = g1; f2 = g2; f3 = g3;
    }

    #pragma unroll
    for (int nt = 0; nt < 16; ++nt) {
        const int col  = nt * 16 + lrow;
        const float bb = bval[col];
        #pragma unroll
        for (int v = 0; v < 8; ++v) {
            const size_t row = (size_t)mtile * 16 + v + khalf * 8;
            val[row * D_ + col] = acc[nt][v] + bb;
        }
    }
}

// ---------------------------------------------------------------------------
// Kernel 3: offsets / softmax weights per (b,q). One 64-thread block per row.
// ---------------------------------------------------------------------------
__global__ __launch_bounds__(64) void proj_kernel(
    const float* __restrict__ pro,
    const float* __restrict__ Woff, const float* __restrict__ boff,
    const float* __restrict__ Wwt,  const float* __restrict__ bwt,
    float* __restrict__ offs, float* __restrict__ wts)
{
    __shared__ float row[D_];
    __shared__ float tmp[H_ * P_];
    const int bq = blockIdx.x;            // 0..1279
    const int t  = threadIdx.x;
    for (int i = t; i < D_; i += 64) row[i] = pro[(size_t)bq * D_ + i];
    __syncthreads();
    if (t < 32) {
        const float* w = Woff + (size_t)t * D_;
        float s = boff[t];
        for (int k = 0; k < D_; ++k) s += row[k] * w[k];
        offs[(size_t)bq * 32 + t] = s;
    } else {
        const int u = t - 32;
        const float* w = Wwt + (size_t)u * D_;
        float s = bwt[u];
        for (int k = 0; k < D_; ++k) s += row[k] * w[k];
        tmp[u] = s;
    }
    __syncthreads();
    if (t < H_) {
        float m = tmp[t * 4];
        for (int p = 1; p < 4; ++p) m = fmaxf(m, tmp[t * 4 + p]);
        float e[4], ssum = 0.f;
        for (int p = 0; p < 4; ++p) { e[p] = __expf(tmp[t * 4 + p] - m); ssum += e[p]; }
        const float inv = 1.0f / ssum;
        for (int p = 0; p < 4; ++p) wts[(size_t)bq * 32 + t * 4 + p] = e[p] * inv;
    }
}

// ---------------------------------------------------------------------------
// Kernel 4: linear-interp gather + weighted sum over P. One wave per (b,q,h),
// lane = dh channel. Writes out_pre[bq][dh*H + h] as bf16 (A of final GEMM).
// ---------------------------------------------------------------------------
__global__ __launch_bounds__(256) void sample_kernel(
    const float* __restrict__ val, const float* __restrict__ offs,
    const float* __restrict__ wts, const float* __restrict__ bpt,
    const float* __restrict__ wsz, __bf16* __restrict__ out_pre)
{
    const int wid = threadIdx.x >> 5;
    const int dh  = threadIdx.x & 31;
    const int gid = blockIdx.x * 8 + wid;     // (b,q,h) flat, 0..10239
    const int b   = gid / (NQ_ * H_);
    const int rem = gid % (NQ_ * H_);
    const int q   = rem / H_;
    const int h   = rem % H_;
    const int bq  = b * NQ_ + q;

    const float bpv = bpt[bq];
    const float w   = wsz[b];
    const float* vb = val + (size_t)b * T_ * D_ + h * DH_ + dh;

    float acc = 0.f;
    #pragma unroll
    for (int p = 0; p < P_; ++p) {
        const int idx  = (bq * H_ + h) * P_ + p;
        const float off = offs[idx];
        const float wt  = wts[idx];
        float sp = (bpv + off / w) * (w / (float)T_);
        sp = fminf(fmaxf(sp, 0.f), 1.f);
        const float x  = sp * (float)(T_ - 1);
        const float x0 = floorf(x);
        const float wx = x - x0;
        int i0 = (int)x0;
        i0 = i0 < 0 ? 0 : (i0 > T_ - 1 ? T_ - 1 : i0);
        const int i1 = i0 + 1 > T_ - 1 ? T_ - 1 : i0 + 1;
        const float v0 = vb[(size_t)i0 * D_];
        const float v1 = vb[(size_t)i1 * D_];
        acc += wt * ((1.f - wx) * v0 + wx * v1);
    }
    out_pre[(size_t)bq * D_ + dh * H_ + h] = (__bf16)acc;
}

// ---------------------------------------------------------------------------
// Kernel 5: out = out_pre @ W_out^T + b_out   (M=1280, K=256, N=256), bf16 WMMA
// ---------------------------------------------------------------------------
__global__ __launch_bounds__(256) void gemm_out(
    const __bf16* __restrict__ apre, const __bf16* __restrict__ wob,
    const float* __restrict__ bout, float* __restrict__ out)
{
    const int lane  = threadIdx.x & 31;
    const int lrow  = lane & 15;
    const int khalf = lane >> 4;
    const int wid   = threadIdx.x >> 5;
    const int mtile = blockIdx.x * 8 + wid;   // 0..79

    const __bf16* arow = apre + ((size_t)mtile * 16 + lrow) * D_;

    v8f acc[16] = {};

    for (int kk = 0; kk < 8; ++kk) {
        const int kb = kk * 32 + khalf * 8;
        v8bf a0 = *(const v8bf*)(arow + kb);
        v8bf a1 = *(const v8bf*)(arow + kb + 16);
        v16bf a;
        #pragma unroll
        for (int i = 0; i < 8; ++i) { a[i] = a0[i]; a[8 + i] = a1[i]; }
        const __bf16* bbase = wob + (size_t)lrow * D_ + kk * 32 + khalf * 8;
        #pragma unroll
        for (int nt = 0; nt < 16; ++nt) {
            const __bf16* bp = bbase + (size_t)nt * 16 * D_;
            v8bf b0 = *(const v8bf*)(bp);
            v8bf b1 = *(const v8bf*)(bp + 16);
            v16bf bfrag;
            #pragma unroll
            for (int i = 0; i < 8; ++i) { bfrag[i] = b0[i]; bfrag[8 + i] = b1[i]; }
            acc[nt] = __builtin_amdgcn_wmma_f32_16x16x32_bf16(
                false, a, false, bfrag, (short)0, acc[nt], false, false);
        }
    }

    #pragma unroll
    for (int nt = 0; nt < 16; ++nt) {
        const int col  = nt * 16 + lrow;
        const float bb = bout[col];
        #pragma unroll
        for (int v = 0; v < 8; ++v) {
            const size_t row = (size_t)mtile * 16 + v + khalf * 8;
            out[row * D_ + col] = acc[nt][v] + bb;
        }
    }
}

// ---------------------------------------------------------------------------
// Host-side launcher
// ---------------------------------------------------------------------------
extern "C" void kernel_launch(void* const* d_in, const int* in_sizes, int n_in,
                              void* d_out, int out_size, void* d_ws, size_t ws_size,
                              hipStream_t stream) {
    (void)in_sizes; (void)n_in; (void)out_size; (void)ws_size;

    const float* pro   = (const float*)d_in[0];   // (32,40,256)
    const float* feats = (const float*)d_in[1];   // (32,4096,512)
    const float* bpt   = (const float*)d_in[2];   // (32,40,1)
    const float* wsz   = (const float*)d_in[3];   // (32,)
    const float* Wval  = (const float*)d_in[4];   // (256,512)
    const float* bval  = (const float*)d_in[5];   // (256,)
    const float* Woff  = (const float*)d_in[6];   // (32,256)
    const float* boff  = (const float*)d_in[7];   // (32,)
    const float* Wwt   = (const float*)d_in[8];   // (32,256)
    const float* bwt   = (const float*)d_in[9];   // (32,)
    const float* Wout  = (const float*)d_in[10];  // (256,256)
    const float* bout  = (const float*)d_in[11];  // (256,)
    float* out = (float*)d_out;                   // (1,1280,256)

    // Workspace layout (bytes)
    char* ws = (char*)d_ws;
    const size_t OFF_VAL = 0;                                   // 32*4096*256 f32 = 128 MiB
    const size_t OFF_WVB = OFF_VAL + (size_t)N_ * T_ * D_ * 4;  // 256*512 bf16
    const size_t OFF_WOB = OFF_WVB + (size_t)D_ * K2D * 2;      // 256*256 bf16
    const size_t OFF_OFS = OFF_WOB + (size_t)D_ * D_ * 2;       // 1280*32 f32
    const size_t OFF_WTS = OFF_OFS + (size_t)N_ * NQ_ * 32 * 4; // 1280*32 f32
    const size_t OFF_PRE = OFF_WTS + (size_t)N_ * NQ_ * 32 * 4; // 1280*256 bf16

    float*  val     = (float*)(ws + OFF_VAL);
    __bf16* wvb     = (__bf16*)(ws + OFF_WVB);
    __bf16* wob     = (__bf16*)(ws + OFF_WOB);
    float*  offs    = (float*)(ws + OFF_OFS);
    float*  wts     = (float*)(ws + OFF_WTS);
    __bf16* out_pre = (__bf16*)(ws + OFF_PRE);

    // 1) weight conversion (covers max(131072, 65536) elems)
    cvt_weights<<<(D_ * K2D + 255) / 256, 256, 0, stream>>>(Wval, Wout, wvb, wob);

    // 2) big GEMM: 131072/16 = 8192 M-tiles, 8 waves/block -> 1024 blocks
    gemm_val<<<1024, 256, 0, stream>>>(feats, wvb, bval, val);

    // 3) projections + softmax: one block per (b,q)
    proj_kernel<<<N_ * NQ_, 64, 0, stream>>>(pro, Woff, boff, Wwt, bwt, offs, wts);

    // 4) sampling: 10240 (b,q,h) waves, 8 waves/block -> 1280 blocks
    sample_kernel<<<(N_ * NQ_ * H_) / 8, 256, 0, stream>>>(val, offs, wts, bpt, wsz, out_pre);

    // 5) output GEMM: 80 M-tiles, 8 waves/block -> 10 blocks
    gemm_out<<<10, 256, 0, stream>>>(out_pre, wob, bout, out);
}
